// GroupedQueryAttention_22273700397073
// MI455X (gfx1250) — matrix operand order repaired and checked by reading, not compiled, not run on hardware
//
#include <hip/hip_runtime.h>
#include <hip/hip_bf16.h>

typedef _Float16 v16h __attribute__((ext_vector_type(16)));
typedef _Float16 v8h  __attribute__((ext_vector_type(8)));
typedef float    v8f  __attribute__((ext_vector_type(8)));

// Problem constants
#define BB   2
#define TT   2048
#define DM   2048
#define NH   16
#define NKV  4
#define DK   128
#define MROWS (BB * TT)          // 4096
#define SOFTMAX_SCALE 0.08838834764831845f  // 1/sqrt(128)

static __device__ __forceinline__ v8h ld8(const _Float16* p) {
    return *(const v8h*)p;
}

static __device__ __forceinline__ v16h pack16(v8h lo, v8h hi) {
    v16h r;
#pragma unroll
    for (int i = 0; i < 8; ++i) { r[i] = lo[i]; r[i + 8] = hi[i]; }
    return r;
}

static __device__ __forceinline__ v8f wmma_f16(v16h a, v16h b, v8f c) {
    // D = A(16x32 f16) x B(32x16 f16) + C(16x16 f32)
    return __builtin_amdgcn_wmma_f32_16x16x32_f16(
        false, a, false, b, (short)0, c, false, false);
}

static __device__ __forceinline__ v8f zero8() {
    v8f z;
#pragma unroll
    for (int i = 0; i < 8; ++i) z[i] = 0.0f;
    return z;
}

// Async copy of one 16B chunk per lane: global -> LDS (ASYNCcnt tracked).
static __device__ __forceinline__ void async_cp16(const _Float16* gsrc,
                                                  _Float16* ldst) {
    unsigned dst = (unsigned)(uintptr_t)ldst;  // low 32 bits = LDS offset
    asm volatile("global_load_async_to_lds_b128 %0, %1, off"
                 :: "v"(dst), "v"(gsrc)
                 : "memory");
}

static __device__ __forceinline__ void wait_async0() {
    asm volatile("s_wait_asynccnt 0x0" ::: "memory");
}

// ---------------------------------------------------------------------------
// Prep kernels
// ---------------------------------------------------------------------------
__global__ void cvt_f32_to_f16(const float* __restrict__ src,
                               _Float16* __restrict__ dst, int n) {
    int i = blockIdx.x * blockDim.x + threadIdx.x;
    if (i < n) dst[i] = (_Float16)src[i];
}

// W [K][N] f32 (row-major) -> Wt [N][K] f16
__global__ void cvt_transpose(const float* __restrict__ W,
                              _Float16* __restrict__ Wt, int K, int N) {
    int i = blockIdx.x * blockDim.x + threadIdx.x;
    if (i < K * N) {
        int k = i / N;
        int n = i - k * N;
        Wt[(size_t)n * K + k] = (_Float16)W[i];
    }
}

// bool mask [B*T*T] -> bit-packed u32 words [B*T*T/32]
__global__ void pack_mask(const unsigned char* __restrict__ m,
                          unsigned int* __restrict__ mb, int nwords) {
    int i = blockIdx.x * blockDim.x + threadIdx.x;
    if (i < nwords) {
        const unsigned char* p = m + (size_t)i * 32;
        unsigned int w = 0;
#pragma unroll
        for (int j = 0; j < 32; ++j) w |= (p[j] ? 1u : 0u) << j;
        mb[i] = w;
    }
}

// ---------------------------------------------------------------------------
// Generic WMMA GEMM with async double-buffered LDS staging:
//   C[M][N] = A[M][K] (f16, row-major) @ Bt[N][K] (f16, transposed) + bias
// Block: 128 threads = 4 waves. Block tile: 256(M) x 64(N).
// Wave tile: 64(M) x 64(N) = 4x4 accumulator tiles.
// K is consumed in 64-wide chunks: A tile (256x64, 32KB) + B tile (64x64,
// 8KB) staged via global_load_async_to_lds_b128, double-buffered (80KB LDS).
// B is staged once per block instead of once per wave (4x traffic cut).
// ---------------------------------------------------------------------------
#define EPI_OUT 0   // f32 row-major [M][N] + bias (final projection)
#define EPI_Q   1   // f16 -> Qh [B][NH][T][DK], scaled by 1/sqrt(DK)
#define EPI_K   2   // f16 -> Kh [B][NKV][T][DK]
#define EPI_VT  3   // f16 -> Vt [B][NKV][DK][T]

template <int EPI>
__global__ __launch_bounds__(128) void wmma_gemm(
    const _Float16* __restrict__ A, const _Float16* __restrict__ Bt,
    const float* __restrict__ bias, void* __restrict__ outp,
    int M, int N, int K) {
    __shared__ _Float16 Abuf[2][256][64];  // 2 x 32KB
    __shared__ _Float16 Bbuf[2][64][64];   // 2 x 8KB

    const int tid  = threadIdx.x;
    const int lane = tid & 31;
    const int wv   = tid >> 5;
    const int g    = lane >> 4;     // half-wave group (0/1)
    const int ln   = lane & 15;     // lane within half
    const int m0blk = blockIdx.y * 256;
    const int mw    = wv * 64;      // wave's rows within block tile
    const int n0    = blockIdx.x * 64;

    // cooperative async stage of one 64-wide K chunk of A and B
    auto stage = [&](int buf, int kk0) {
        _Float16* adst = &Abuf[buf][0][0];
#pragma unroll
        for (int s = 0; s < 16; ++s) {
            const int c = s * 128 + tid;          // 16B chunk id, 0..2047
            const int row = c >> 3, cc = c & 7;   // 8 chunks per 128B row
            async_cp16(A + (size_t)(m0blk + row) * K + kk0 + cc * 8,
                       adst + c * 8);
        }
        _Float16* bdst = &Bbuf[buf][0][0];
#pragma unroll
        for (int s = 0; s < 4; ++s) {
            const int c = s * 128 + tid;          // 16B chunk id, 0..511
            const int row = c >> 3, cc = c & 7;
            async_cp16(Bt + (size_t)(n0 + row) * K + kk0 + cc * 8,
                       bdst + c * 8);
        }
    };

    v8f c[4][4];
#pragma unroll
    for (int i = 0; i < 4; ++i)
#pragma unroll
        for (int j = 0; j < 4; ++j) c[i][j] = zero8();

    stage(0, 0);  // prologue: fill buffer 0

    const int nk64 = K >> 6;
    for (int t = 0; t < nk64; ++t) {
        const int cur = t & 1;
        wait_async0();    // this wave's staging for `cur` complete
        __syncthreads();  // => everyone's staging visible

        if (t + 1 < nk64) stage(1 - cur, (t + 1) * 64);  // overlap next chunk

#pragma unroll
        for (int ks = 0; ks < 2; ++ks) {  // two 32-wide sub-steps
            v16h a[4];
#pragma unroll
            for (int i = 0; i < 4; ++i) {
                const _Float16* p = &Abuf[cur][mw + i * 16 + ln][ks * 32 + 8 * g];
                a[i] = pack16(ld8(p), ld8(p + 16));
            }
            v16h b[4];
#pragma unroll
            for (int j = 0; j < 4; ++j) {
                const _Float16* p = &Bbuf[cur][j * 16 + ln][ks * 32 + 8 * g];
                b[j] = pack16(ld8(p), ld8(p + 16));
            }
#pragma unroll
            for (int i = 0; i < 4; ++i)
#pragma unroll
                for (int j = 0; j < 4; ++j) c[i][j] = wmma_f16(a[i], b[j], c[i][j]);
        }

        __syncthreads();  // all waves done reading `cur` before restage
    }

    // Epilogue. C/D layout: row M = r + 8*g, col N = ln, in VGPR r.
#pragma unroll
    for (int i = 0; i < 4; ++i) {
#pragma unroll
        for (int j = 0; j < 4; ++j) {
            const int col = n0 + j * 16 + ln;
            const float bv = bias[col];
#pragma unroll
            for (int r = 0; r < 8; ++r) {
                const int m = m0blk + mw + i * 16 + r + 8 * g;
                float val = c[i][j][r] + bv;
                if (EPI == EPI_OUT) {
                    ((float*)outp)[(size_t)m * N + col] = val;
                } else {
                    const int bidx = m >> 11;       // m / TT
                    const int t    = m & (TT - 1);  // m % TT
                    const int hh   = col >> 7;      // col / DK
                    const int d    = col & (DK - 1);
                    _Float16* o16 = (_Float16*)outp;
                    if (EPI == EPI_Q) {
                        val *= SOFTMAX_SCALE;  // fold 1/sqrt(d) into Q
                        o16[(((size_t)bidx * NH + hh) * TT + t) * DK + d] = (_Float16)val;
                    } else if (EPI == EPI_K) {
                        o16[(((size_t)bidx * NKV + hh) * TT + t) * DK + d] = (_Float16)val;
                    } else { // EPI_VT: transposed V [B][NKV][DK][T]
                        o16[(((size_t)bidx * NKV + hh) * DK + d) * TT + t] = (_Float16)val;
                    }
                }
            }
        }
    }
}

// ---------------------------------------------------------------------------
// Flash-style GQA attention with async double-buffered K/V staging in LDS.
// One wave owns 16 queries x full DK=128, streaming keys in blocks of 64.
// Block = 128 threads (4 waves) sharing (b,h); all waves consume the same
// K/V blocks, staged once per block via global_load_async_to_lds_b128.
// Grid = (T/64, NH, B).
// ---------------------------------------------------------------------------
__global__ __launch_bounds__(128) void gqa_attention(
    const _Float16* __restrict__ Qh,   // [B][NH][T][DK] (pre-scaled by 1/sqrt(d))
    const _Float16* __restrict__ Kh,   // [B][NKV][T][DK]
    const _Float16* __restrict__ Vt,   // [B][NKV][DK][T]
    const unsigned int* __restrict__ maskb,  // bit-packed [B][T][T/32]
    _Float16* __restrict__ Oh)         // [B*T][DM] (token-major, head-packed)
{
    __shared__ _Float16 Kbuf[2][64][DK];   // 2 x 16KB: 64 keys x 128 d
    __shared__ _Float16 Vbuf[2][DK][64];   // 2 x 16KB: 128 d x 64 keys
    __shared__ _Float16 plds[4][16][64];   // per-wave P staging (16q x 64k)

    const int tid  = threadIdx.x;
    const int lane = tid & 31;
    const int wv   = tid >> 5;
    const int g    = lane >> 4;
    const int ln   = lane & 15;

    const int b  = blockIdx.z;
    const int h  = blockIdx.y;
    const int hk = h >> 2;  // NH/NKV = 4 query heads per KV head
    const int qbase = blockIdx.x * 64 + wv * 16;

    const _Float16* Qp = Qh + ((size_t)b * NH + h) * TT * DK;
    const _Float16* Kp = Kh + ((size_t)b * NKV + hk) * TT * DK;
    const _Float16* Vp = Vt + ((size_t)b * NKV + hk) * DK * TT;
    const unsigned int* mbp = maskb + (size_t)b * TT * (TT / 32);

    // cooperative async stage of one 64-key K/V block into LDS buffer `buf`
    auto stage_kv = [&](int buf, int jb) {
        // K block: rows jb..jb+63 are contiguous 16KB in Kh -> linear copy.
        const _Float16* ksrc = Kp + (size_t)jb * DK;
        _Float16* kdst = &Kbuf[buf][0][0];
#pragma unroll
        for (int s = 0; s < 8; ++s) {
            const int c = s * 128 + tid;            // 16B chunk id, 0..1023
            async_cp16(ksrc + c * 8, kdst + c * 8);
        }
        // V block: 128 rows (d), 128B from each row at column jb.
        _Float16* vdst = &Vbuf[buf][0][0];
#pragma unroll
        for (int s = 0; s < 8; ++s) {
            const int c = s * 128 + tid;            // 16B chunk id, 0..1023
            const int row = c >> 3, cc = c & 7;
            async_cp16(Vp + (size_t)row * TT + jb + cc * 8, vdst + c * 8);
        }
    };

    // Load Q fragments for the full d=128 (4 k-chunks of 32), keep resident.
    v16h qa[4];
#pragma unroll
    for (int kk = 0; kk < 4; ++kk) {
        const _Float16* p = Qp + (size_t)(qbase + ln) * DK + kk * 32 + 8 * g;
        qa[kk] = pack16(ld8(p), ld8(p + 16));
    }

    // all-ones B fragment for WMMA row sums
    v16h onesb;
#pragma unroll
    for (int i = 0; i < 16; ++i) onesb[i] = (_Float16)1.0f;

    float mrun[8], lrun[8];
    v8f o[8];
#pragma unroll
    for (int r = 0; r < 8; ++r) { mrun[r] = -1e30f; lrun[r] = 0.0f; }
#pragma unroll
    for (int dt = 0; dt < 8; ++dt) o[dt] = zero8();

    stage_kv(0, 0);  // prologue: fill buffer 0

    for (int ib = 0; ib < TT / 64; ++ib) {
        const int jb = ib * 64;
        const int cur = ib & 1;

        wait_async0();    // this wave's staging for `cur` is complete
        __syncthreads();  // => everyone's staging for `cur` is visible

        if (ib + 1 < TT / 64) stage_kv(1 - cur, jb + 64);  // overlap next

        // ---- scores: S[16q x 64k] = Q @ K^T (four 16x16 f32 tiles) ----
        v8f s[4];
#pragma unroll
        for (int u = 0; u < 4; ++u) s[u] = zero8();
#pragma unroll
        for (int kk = 0; kk < 4; ++kk) {
            v16h bk[4];
#pragma unroll
            for (int u = 0; u < 4; ++u) {
                const _Float16* p = &Kbuf[cur][u * 16 + ln][kk * 32 + 8 * g];
                bk[u] = pack16(ld8(p), ld8(p + 16));
            }
#pragma unroll
            for (int u = 0; u < 4; ++u) s[u] = wmma_f16(qa[kk], bk[u], s[u]);
        }

        // ---- online softmax update (bit-packed mask) ----
        float alpha[8];
#pragma unroll
        for (int r = 0; r < 8; ++r) {
            const int q = qbase + r + 8 * g;
            const unsigned int* mw = mbp + (size_t)q * (TT / 32) + (jb >> 5);
            const unsigned int w0 = mw[0];  // keys jb .. jb+31
            const unsigned int w1 = mw[1];  // keys jb+32 .. jb+63
            float x0 = ((w0 >> ln) & 1u)        ? s[0][r] : -1e9f;
            float x1 = ((w0 >> (ln + 16)) & 1u) ? s[1][r] : -1e9f;
            float x2 = ((w1 >> ln) & 1u)        ? s[2][r] : -1e9f;
            float x3 = ((w1 >> (ln + 16)) & 1u) ? s[3][r] : -1e9f;
            // row max across the 16 lanes of this half-wave
            float t = fmaxf(fmaxf(x0, x1), fmaxf(x2, x3));
            t = fmaxf(t, __shfl_xor(t, 1, 32));
            t = fmaxf(t, __shfl_xor(t, 2, 32));
            t = fmaxf(t, __shfl_xor(t, 4, 32));
            t = fmaxf(t, __shfl_xor(t, 8, 32));
            const float mnew = fmaxf(mrun[r], t);
            alpha[r] = __expf(mrun[r] - mnew);
            mrun[r] = mnew;
            const float p0 = __expf(x0 - mnew);
            const float p1 = __expf(x1 - mnew);
            const float p2 = __expf(x2 - mnew);
            const float p3 = __expf(x3 - mnew);
            // stage P (C layout) into LDS row-major [16][64]
            _Float16* row = &plds[wv][r + 8 * g][0];
            row[ln]      = (_Float16)p0;
            row[16 + ln] = (_Float16)p1;
            row[32 + ln] = (_Float16)p2;
            row[48 + ln] = (_Float16)p3;
        }
#pragma unroll
        for (int dt = 0; dt < 8; ++dt)
#pragma unroll
            for (int r = 0; r < 8; ++r) o[dt][r] *= alpha[r];

        // make cross-lane LDS writes visible (wave-private buffer)
        asm volatile("s_wait_dscnt 0" ::: "memory");

        // reload P as two A fragments (16x32 each)
        const _Float16* pp = &plds[wv][ln][0];
        v16h pa0 = pack16(ld8(pp + 8 * g),      ld8(pp + 16 + 8 * g));
        v16h pa1 = pack16(ld8(pp + 32 + 8 * g), ld8(pp + 48 + 8 * g));

        // row sums of P via WMMA against all-ones B (lands in C layout)
        v8f sumt = wmma_f16(pa0, onesb, zero8());
        sumt = wmma_f16(pa1, onesb, sumt);
#pragma unroll
        for (int r = 0; r < 8; ++r) lrun[r] = lrun[r] * alpha[r] + sumt[r];

        // ---- O += P @ V  (V from LDS, transposed [DK][64-key block]) ----
#pragma unroll
        for (int half = 0; half < 2; ++half) {
            const v16h pa = half ? pa1 : pa0;
#pragma unroll
            for (int dq = 0; dq < 2; ++dq) {
                v16h bv[4];
#pragma unroll
                for (int u = 0; u < 4; ++u) {
                    const int dt = dq * 4 + u;
                    const _Float16* pv =
                        &Vbuf[cur][dt * 16 + ln][32 * half + 8 * g];
                    bv[u] = pack16(ld8(pv), ld8(pv + 16));
                }
#pragma unroll
                for (int u = 0; u < 4; ++u) {
                    const int dt = dq * 4 + u;
                    o[dt] = wmma_f16(pa, bv[u], o[dt]);
                }
            }
        }

        __syncthreads();  // all waves done reading `cur` before it is restaged
    }

    // ---- normalize and store to Oh [b*T + q][h*DK + d] ----
    float inv[8];
#pragma unroll
    for (int r = 0; r < 8; ++r) inv[r] = 1.0f / lrun[r];
#pragma unroll
    for (int dt = 0; dt < 8; ++dt) {
#pragma unroll
        for (int r = 0; r < 8; ++r) {
            const int q = qbase + r + 8 * g;
            const size_t idx =
                ((size_t)b * TT + q) * DM + h * DK + dt * 16 + ln;
            Oh[idx] = (_Float16)(o[dt][r] * inv[r]);
        }
    }
}

// ---------------------------------------------------------------------------
// Host-side launch
// ---------------------------------------------------------------------------
extern "C" void kernel_launch(void* const* d_in, const int* in_sizes, int n_in,
                              void* d_out, int out_size, void* d_ws, size_t ws_size,
                              hipStream_t stream) {
    const float* x    = (const float*)d_in[0];
    const unsigned char* mask = (const unsigned char*)d_in[1];
    const float* Wq   = (const float*)d_in[2];
    const float* bq   = (const float*)d_in[3];
    const float* Wk   = (const float*)d_in[4];
    const float* bk   = (const float*)d_in[5];
    const float* Wv   = (const float*)d_in[6];
    const float* bv   = (const float*)d_in[7];
    const float* Wo   = (const float*)d_in[8];
    const float* bo   = (const float*)d_in[9];
    float* out        = (float*)d_out;

    // Workspace layout, 256B aligned slices
    char* ws = (char*)d_ws;
    size_t off = 0;
    auto takeb = [&](size_t bytes) {
        size_t o = off;
        off = (off + bytes + 255) & ~(size_t)255;
        return (void*)(ws + o);
    };
    auto take = [&](size_t halves) {
        return (_Float16*)takeb(halves * sizeof(_Float16));
    };
    _Float16* xh  = take((size_t)MROWS * DM);        // x f16
    _Float16* Wqt = take((size_t)DM * DM);           // Wq^T [N][K]
    _Float16* Wkt = take((size_t)(NKV * DK) * DM);   // Wk^T
    _Float16* Wvt = take((size_t)(NKV * DK) * DM);   // Wv^T
    _Float16* Wot = take((size_t)DM * DM);           // Wo^T
    _Float16* Qh  = take((size_t)BB * NH * TT * DK);
    _Float16* Kh  = take((size_t)BB * NKV * TT * DK);
    _Float16* Vt  = take((size_t)BB * NKV * DK * TT);
    _Float16* Oh  = take((size_t)MROWS * DM);
    unsigned int* mb = (unsigned int*)takeb((size_t)BB * TT * (TT / 32) * 4);

    // 1) conversions / packing
    {
        int n = MROWS * DM;
        cvt_f32_to_f16<<<(n + 255) / 256, 256, 0, stream>>>(x, xh, n);
    }
    cvt_transpose<<<(DM * DM + 255) / 256, 256, 0, stream>>>(Wq, Wqt, DM, DM);
    cvt_transpose<<<(DM * NKV * DK + 255) / 256, 256, 0, stream>>>(Wk, Wkt, DM, NKV * DK);
    cvt_transpose<<<(DM * NKV * DK + 255) / 256, 256, 0, stream>>>(Wv, Wvt, DM, NKV * DK);
    cvt_transpose<<<(DM * DM + 255) / 256, 256, 0, stream>>>(Wo, Wot, DM, DM);
    {
        int nwords = BB * TT * (TT / 32);
        pack_mask<<<(nwords + 255) / 256, 256, 0, stream>>>(mask, mb, nwords);
    }

    // 2) QKV projections (WMMA GEMMs with layout-scatter epilogues)
    {
        dim3 blk(128);
        dim3 gq(DM / 64, MROWS / 256);
        wmma_gemm<EPI_Q><<<gq, blk, 0, stream>>>(xh, Wqt, bq, Qh, MROWS, DM, DM);
        dim3 gkv((NKV * DK) / 64, MROWS / 256);
        wmma_gemm<EPI_K><<<gkv, blk, 0, stream>>>(xh, Wkt, bk, Kh, MROWS, NKV * DK, DM);
        wmma_gemm<EPI_VT><<<gkv, blk, 0, stream>>>(xh, Wvt, bv, Vt, MROWS, NKV * DK, DM);
    }

    // 3) attention
    {
        dim3 blk(128);
        dim3 grd(TT / 64, NH, BB);
        gqa_attention<<<grd, blk, 0, stream>>>(Qh, Kh, Vt, mb, Oh);
    }

    // 4) output projection -> f32
    {
        dim3 blk(128);
        dim3 go(DM / 64, MROWS / 256);
        wmma_gemm<EPI_OUT><<<go, blk, 0, stream>>>(Oh, Wot, bo, out, MROWS, DM, DM);
    }
}
